// PLELayer_86955907875104
// MI455X (gfx1250) — matrix operand order repaired and checked by reading, not compiled, not run on hardware
//
#include <hip/hip_runtime.h>

typedef unsigned short u16;
typedef unsigned int   u32;
typedef __attribute__((ext_vector_type(16))) __bf16 v16bf;
typedef __attribute__((ext_vector_type(8)))  float  v8f;

struct __align__(16) u32x4 { u32 x, y, z, w; };
struct u32x8 { u32x4 lo, hi; };
struct __align__(16) f32x4 { float x, y, z, w; };

__device__ __forceinline__ u16 f2bf(float f) {
  u32 u = __builtin_bit_cast(u32, f);
  u32 r = u + 0x7FFFu + ((u >> 16) & 1u);
  return (u16)(r >> 16);
}

// Build a 16-element bf16 fragment from two 16-byte chunks.
__device__ __forceinline__ v16bf ld_frag(const u16* p0, const u16* p1) {
  u32x8 u;
  u.lo = *(const u32x4*)p0;
  u.hi = *(const u32x4*)p1;
  return __builtin_bit_cast(v16bf, u);
}

__device__ __forceinline__ v8f wmma_bf16(v16bf a, v16bf b, v8f c) {
  return __builtin_amdgcn_wmma_f32_16x16x32_bf16(false, a, false, b, (short)0, c,
                                                 false, false);
}

__device__ __forceinline__ v8f vzero8() {
  v8f z = {0.f, 0.f, 0.f, 0.f, 0.f, 0.f, 0.f, 0.f};
  return z;
}

// CDNA5 async copy global -> LDS (16B per lane), tracked by ASYNCcnt.
// Generic LDS pointers carry the LDS byte offset in addr[31:0] (ISA 10.2).
__device__ __forceinline__ void async_copy_b128(void* lds_dst, const void* gsrc) {
  u32 l = (u32)(size_t)lds_dst;
  asm volatile("global_load_async_to_lds_b128 %0, %1, off"
               :: "v"(l), "v"(gsrc) : "memory");
}
__device__ __forceinline__ void wait_async0() {
  asm volatile("s_wait_asynccnt 0x0" ::: "memory");
}

// ---------------- Stage 0a: x fp32 -> bf16 ----------------
__global__ void __launch_bounds__(256) k_cvtx(const float* __restrict__ x,
                                              u16* __restrict__ xb) {
  size_t base = ((size_t)blockIdx.x * 256 + threadIdx.x) * 8;
  f32x4 a = *(const f32x4*)(x + base);
  f32x4 b = *(const f32x4*)(x + base + 4);
  u32x4 o;
  o.x = (u32)f2bf(a.x) | ((u32)f2bf(a.y) << 16);
  o.y = (u32)f2bf(a.z) | ((u32)f2bf(a.w) << 16);
  o.z = (u32)f2bf(b.x) | ((u32)f2bf(b.y) << 16);
  o.w = (u32)f2bf(b.z) | ((u32)f2bf(b.w) << 16);
  *(u32x4*)(xb + base) = o;
}

// ---------------- Stage 0b: weight [K][N] fp32 -> [N][K] bf16 ----------------
__global__ void __launch_bounds__(256) k_tcvt(const float* __restrict__ in,
                                              u16* __restrict__ out, int K, int N) {
  __shared__ float tile[32][33];
  const int tid = threadIdx.x, tx = tid & 31, ty = tid >> 5;
  const size_t mat = (size_t)blockIdx.z * K * N;
  const int k0 = blockIdx.x * 32, n0 = blockIdx.y * 32;
#pragma unroll
  for (int r = 0; r < 4; ++r)
    tile[ty + r * 8][tx] = in[mat + (size_t)(k0 + ty + r * 8) * N + n0 + tx];
  __syncthreads();
#pragma unroll
  for (int r = 0; r < 4; ++r)
    out[mat + (size_t)(n0 + ty + r * 8) * K + k0 + tx] = f2bf(tile[tx][ty + r * 8]);
}

// ---------------- Stage 1: gates [3][B][6] ----------------
__global__ void __launch_bounds__(256) k_gates(const u16* __restrict__ xb,
                                               const u16* __restrict__ gw1t,
                                               const float* __restrict__ gb1,
                                               const float* __restrict__ gW2,
                                               const float* __restrict__ gb2,
                                               float* __restrict__ gates) {
  __shared__ float gh[64 * 64];  // 16 KB
  const int tid = threadIdx.x, lane = tid & 31, w = tid >> 5;
  const int l16 = lane & 15, lhi = lane >> 4;
  const int mw = w >> 1, nw = w & 1;  // 4 x 2 waves: M=16, N=32 per wave
  const int mblk = blockIdx.x * 64;
  const int cA = lhi * 8, cB = lhi * 8;

#pragma unroll 1
  for (int g = 0; g < 3; ++g) {
    v8f acc[2];
    acc[0] = vzero8(); acc[1] = vzero8();
    const u16* ar = xb + (size_t)(mblk + mw * 16 + l16) * 1024;
    const u16* br0 = gw1t + ((size_t)g * 64 + nw * 32 + l16) * 1024;
    const u16* br1 = gw1t + ((size_t)g * 64 + nw * 32 + 16 + l16) * 1024;
#pragma unroll 2
    for (int k = 0; k < 1024; k += 32) {
      v16bf a  = ld_frag(ar + k + cA, ar + k + cA + 16);
      v16bf b0 = ld_frag(br0 + k + cB, br0 + k + cB + 16);
      v16bf b1 = ld_frag(br1 + k + cB, br1 + k + cB + 16);
      acc[0] = wmma_bf16(a, b0, acc[0]);
      acc[1] = wmma_bf16(a, b1, acc[1]);
    }
#pragma unroll
    for (int nt = 0; nt < 2; ++nt) {
      int n = nw * 32 + nt * 16 + l16;
      float bv = gb1[g * 64 + n];
#pragma unroll
      for (int i = 0; i < 8; ++i) {
        int m = mw * 16 + i + lhi * 8;
        gh[m * 64 + n] = fmaxf(acc[nt][i] + bv, 0.f);
      }
    }
    __syncthreads();
    if (tid < 64) {
      float lg[6];
#pragma unroll
      for (int e6 = 0; e6 < 6; ++e6) {
        float s = gb2[g * 6 + e6];
        for (int k = 0; k < 64; ++k)
          s += gh[tid * 64 + k] * gW2[((size_t)g * 64 + k) * 6 + e6];
        lg[e6] = s;
      }
      float mx = lg[0];
#pragma unroll
      for (int e6 = 1; e6 < 6; ++e6) mx = fmaxf(mx, lg[e6]);
      float ex[6], sm = 0.f;
#pragma unroll
      for (int e6 = 0; e6 < 6; ++e6) { ex[e6] = __expf(lg[e6] - mx); sm += ex[e6]; }
      float inv = 1.f / sm;
#pragma unroll
      for (int e6 = 0; e6 < 6; ++e6)
        gates[((size_t)g * 16384 + mblk + tid) * 6 + e6] = ex[e6] * inv;
    }
    __syncthreads();
  }
}

// ---------------- Stage 2: fused experts + combine (double-buffered) ----------
// LDS: xs 132096 + wsb 139264 + hbuf 33792 + gl 4608 = 309,760 B (< 320 KB)
__global__ void __launch_bounds__(256) k_main(const u16* __restrict__ xb,
                                              const u16* __restrict__ w1t,
                                              const u16* __restrict__ w2t,
                                              const float* __restrict__ gates,
                                              const float* __restrict__ sb1,
                                              const float* __restrict__ db1,
                                              const float* __restrict__ sb2,
                                              const float* __restrict__ db2,
                                              float* __restrict__ out) {
  __shared__ __align__(16) u16 xs[64 * 1032];     // x tile, padded rows
  __shared__ __align__(16) u16 wsb[2][256 * 136]; // double-buffered W1 chunk / W2
  __shared__ __align__(16) u16 hbuf[64 * 264];    // bf16 h tile
  __shared__ float gl[3 * 64 * 6];                // gate tile
  const int tid = threadIdx.x, lane = tid & 31, w = tid >> 5;
  const int mblk = blockIdx.x * 64;
  const int l16 = lane & 15, lhi = lane >> 4;
  const int mw = w >> 2, nw = w & 3;  // 2 x 4 waves
  const int cA = lhi * 8, cB = lhi * 8;

  // Stage W1 chunk (256 rows x 128 k bf16) for expert e_ at k-offset kc_ -> buf p_
  auto stage_w1 = [&](int e_, int kc_, int p_) {
#pragma unroll 4
    for (int r = 0; r < 16; ++r) {
      int idx = tid + r * 256;            // 4096 x 16B chunks
      int row = idx >> 4, c8 = idx & 15;
      async_copy_b128(&wsb[p_][row * 136 + c8 * 8],
                      w1t + ((size_t)e_ * 256 + row) * 1024 + kc_ + c8 * 8);
    }
  };
  // Stage W2 (128 rows x 256 k): k 0..127 -> buf0, k 128..255 -> buf1
  auto stage_w2 = [&](int e_) {
#pragma unroll 4
    for (int r = 0; r < 16; ++r) {
      int idx = tid + r * 256;            // 4096 x 16B chunks
      int h = idx >> 11, rem = idx & 2047;
      int row = rem >> 4, c8 = rem & 15;
      async_copy_b128(&wsb[h][row * 136 + c8 * 8],
                      w2t + ((size_t)e_ * 128 + row) * 256 + h * 128 + c8 * 8);
    }
  };

  // Stage the whole x tile (64 x 1024 bf16) into LDS asynchronously.
  {
    const u16* src = xb + (size_t)mblk * 1024;
#pragma unroll 4
    for (int r = 0; r < 32; ++r) {
      int idx = tid + r * 256;              // 8192 x 16B chunks
      int row = idx >> 7, c8 = idx & 127;
      async_copy_b128(&xs[row * 1032 + c8 * 8], src + row * 1024 + c8 * 8);
    }
  }
  for (int i = tid; i < 3 * 64 * 6; i += 256) {
    int g = i / 384, r = i % 384;
    gl[i] = gates[((size_t)g * 16384 + mblk + r / 6) * 6 + (r % 6)];
  }

  v8f outacc[3][2][2];
#pragma unroll
  for (int g = 0; g < 3; ++g)
#pragma unroll
    for (int a = 0; a < 2; ++a)
#pragma unroll
      for (int b = 0; b < 2; ++b) outacc[g][a][b] = vzero8();

  const u16* arl[2];
  const u16* hrl[2];
#pragma unroll
  for (int mt = 0; mt < 2; ++mt) {
    arl[mt] = xs + (mw * 32 + mt * 16 + l16) * 1032;
    hrl[mt] = hbuf + (mw * 32 + mt * 16 + l16) * 264;
  }
  const u16* brl0[4];
  const u16* brl1[4];
#pragma unroll
  for (int nt = 0; nt < 4; ++nt) {
    brl0[nt] = wsb[0] + (nw * 64 + nt * 16 + l16) * 136;
    brl1[nt] = wsb[1] + (nw * 64 + nt * 16 + l16) * 136;
  }
  const u16* b2l[2][2];
#pragma unroll
  for (int h = 0; h < 2; ++h)
#pragma unroll
    for (int nt = 0; nt < 2; ++nt)
      b2l[h][nt] = wsb[h] + (nw * 32 + nt * 16 + l16) * 136;

  v8f acc[2][4];
  // One 128-wide K chunk of GEMM1 from buffer `brl` at global k base kb.
  auto gemm1_chunk = [&](int kb, const u16* const* brl) {
#pragma unroll
    for (int kk = 0; kk < 128; kk += 32) {
      int ka = kb + kk;
      v16bf a0 = ld_frag(arl[0] + ka + cA, arl[0] + ka + cA + 16);
      v16bf a1 = ld_frag(arl[1] + ka + cA, arl[1] + ka + cA + 16);
      v16bf bf[4];
#pragma unroll
      for (int nt = 0; nt < 4; ++nt)
        bf[nt] = ld_frag(brl[nt] + kk + cB, brl[nt] + kk + cB + 16);
#pragma unroll
      for (int nt = 0; nt < 4; ++nt) {
        acc[0][nt] = wmma_bf16(a0, bf[nt], acc[0][nt]);
        acc[1][nt] = wmma_bf16(a1, bf[nt], acc[1][nt]);
      }
    }
  };

  stage_w1(0, 0, 0);  // prime the pipeline: expert 0, chunk 0 -> buf0

#pragma unroll 1
  for (int e = 0; e < 10; ++e) {
    // ---- GEMM1: M=64 x N=256 x K=1024, 8 chunks, double-buffered ----
#pragma unroll
    for (int mt = 0; mt < 2; ++mt)
#pragma unroll
      for (int nt = 0; nt < 4; ++nt) acc[mt][nt] = vzero8();

#pragma unroll 1
    for (int c2 = 0; c2 < 8; c2 += 2) {
      // chunk c2 from buf0; stream chunk c2+1 into buf1 meanwhile
      wait_async0();
      __syncthreads();
      stage_w1(e, (c2 + 1) * 128, 1);
      gemm1_chunk(c2 * 128, brl0);
      // chunk c2+1 from buf1; stream chunk c2+2 into buf0 meanwhile
      wait_async0();
      __syncthreads();
      if (c2 + 2 < 8) stage_w1(e, (c2 + 2) * 128, 0);
      gemm1_chunk((c2 + 1) * 128, brl1);
    }
    __syncthreads();  // all waves done with buf1 (last chunk)

    // stream W2 into both buffers while doing the GEMM1 epilogue
    stage_w2(e);
    const float* b1p = (e < 4) ? (sb1 + e * 256) : (db1 + (e - 4) * 256);
#pragma unroll
    for (int mt = 0; mt < 2; ++mt)
#pragma unroll
      for (int nt = 0; nt < 4; ++nt) {
        int n = nw * 64 + nt * 16 + l16;
        float bv = b1p[n];
#pragma unroll
        for (int i = 0; i < 8; ++i) {
          int m = mw * 32 + mt * 16 + i + lhi * 8;
          hbuf[m * 264 + n] = f2bf(fmaxf(acc[mt][nt][i] + bv, 0.f));
        }
      }
    wait_async0();
    __syncthreads();  // hbuf written + W2 staged

    // ---- GEMM2: M=64 x N=128 x K=256, fully LDS-fed ----
    v8f acc2[2][2];
#pragma unroll
    for (int mt = 0; mt < 2; ++mt)
#pragma unroll
      for (int nt = 0; nt < 2; ++nt) acc2[mt][nt] = vzero8();
#pragma unroll
    for (int k = 0; k < 256; k += 32) {
      const int h = k >> 7, kk = k & 127;
      v16bf a0 = ld_frag(hrl[0] + k + cA, hrl[0] + k + cA + 16);
      v16bf a1 = ld_frag(hrl[1] + k + cA, hrl[1] + k + cA + 16);
      v16bf b0 = ld_frag(b2l[h][0] + kk + cB, b2l[h][0] + kk + cB + 16);
      v16bf b1 = ld_frag(b2l[h][1] + kk + cB, b2l[h][1] + kk + cB + 16);
      acc2[0][0] = wmma_bf16(a0, b0, acc2[0][0]);
      acc2[0][1] = wmma_bf16(a0, b1, acc2[0][1]);
      acc2[1][0] = wmma_bf16(a1, b0, acc2[1][0]);
      acc2[1][1] = wmma_bf16(a1, b1, acc2[1][1]);
    }
    __syncthreads();  // all waves done reading W2 from wsb

    // overlap: stream next expert's first W1 chunk during the combine tail
    if (e < 9) {
      stage_w1(e + 1, 0, 0);
      __builtin_prefetch(w1t + ((size_t)(e + 1) * 256) * 1024 + 128, 0, 1);
    }

    const float* b2p = (e < 4) ? (sb2 + e * 128) : (db2 + (e - 4) * 128);
#pragma unroll
    for (int mt = 0; mt < 2; ++mt)
#pragma unroll
      for (int nt = 0; nt < 2; ++nt) {
        int n = nw * 32 + nt * 16 + l16;
        float bv = b2p[n];
#pragma unroll
        for (int i = 0; i < 8; ++i) {
          int m = mw * 32 + mt * 16 + i + lhi * 8;
          float o = fmaxf(acc2[mt][nt][i] + bv, 0.f);
          float w0, w1c, w2c;
          if (e < 4) {
            w0  = gl[m * 6 + e];
            w1c = gl[(64 + m) * 6 + e];
            w2c = gl[(128 + m) * 6 + e];
          } else {
            int t = e - 4, gd = t >> 1, idx = 4 + (t & 1);
            float wv = gl[(gd * 64 + m) * 6 + idx];
            w0  = (gd == 0) ? wv : 0.f;
            w1c = (gd == 1) ? wv : 0.f;
            w2c = (gd == 2) ? wv : 0.f;
          }
          outacc[0][mt][nt][i] += w0 * o;
          outacc[1][mt][nt][i] += w1c * o;
          outacc[2][mt][nt][i] += w2c * o;
        }
      }
  }

  // ---- store out [3][B][128] ----
#pragma unroll
  for (int g = 0; g < 3; ++g)
#pragma unroll
    for (int mt = 0; mt < 2; ++mt)
#pragma unroll
      for (int nt = 0; nt < 2; ++nt) {
        int n = nw * 32 + nt * 16 + l16;
#pragma unroll
        for (int i = 0; i < 8; ++i) {
          int m = mblk + mw * 32 + mt * 16 + i + lhi * 8;
          out[((size_t)g * 16384 + m) * 128 + n] = outacc[g][mt][nt][i];
        }
      }
}

extern "C" void kernel_launch(void* const* d_in, const int* in_sizes, int n_in,
                              void* d_out, int out_size, void* d_ws, size_t ws_size,
                              hipStream_t stream) {
  const float* x   = (const float*)d_in[0];
  // d_in[1] = domain_ids: unused by the reference computation
  const float* sW1 = (const float*)d_in[2];
  const float* sb1 = (const float*)d_in[3];
  const float* sW2 = (const float*)d_in[4];
  const float* sb2 = (const float*)d_in[5];
  const float* dW1 = (const float*)d_in[6];
  const float* db1 = (const float*)d_in[7];
  const float* dW2 = (const float*)d_in[8];
  const float* db2 = (const float*)d_in[9];
  const float* gW1 = (const float*)d_in[10];
  const float* gb1 = (const float*)d_in[11];
  const float* gW2 = (const float*)d_in[12];
  const float* gb2 = (const float*)d_in[13];
  float* out = (float*)d_out;

  char* ws = (char*)d_ws;
  u16* xb      = (u16*)(ws);                    // 16384*1024*2      = 33,554,432
  u16* w1t     = (u16*)(ws + 33554432);         // 10*256*1024*2     =  5,242,880
  u16* w2t     = (u16*)(ws + 38797312);         // 10*128*256*2      =    655,360
  u16* gw1t    = (u16*)(ws + 39452672);         // 3*64*1024*2       =    393,216
  float* gates = (float*)(ws + 39845888);       // 3*16384*6*4       =  1,179,648

  k_cvtx<<<8192, 256, 0, stream>>>(x, xb);
  k_tcvt<<<dim3(32, 8, 4), 256, 0, stream>>>(sW1, w1t, 1024, 256);
  k_tcvt<<<dim3(32, 8, 6), 256, 0, stream>>>(dW1, w1t + (size_t)4 * 256 * 1024, 1024, 256);
  k_tcvt<<<dim3(8, 4, 4), 256, 0, stream>>>(sW2, w2t, 256, 128);
  k_tcvt<<<dim3(8, 4, 6), 256, 0, stream>>>(dW2, w2t + (size_t)4 * 128 * 256, 256, 128);
  k_tcvt<<<dim3(32, 2, 3), 256, 0, stream>>>(gW1, gw1t, 1024, 64);
  k_gates<<<256, 256, 0, stream>>>(xb, gw1t, gb1, gW2, gb2, gates);
  k_main<<<256, 256, 0, stream>>>(xb, w1t, w2t, gates, sb1, db1, sb2, db2, out);
}
